// Self_Attn_suvey_58609123721846
// MI455X (gfx1250) — matrix operand order repaired and checked by reading, not compiled, not run on hardware
//
#include <hip/hip_runtime.h>
#include <hip/hip_bf16.h>
#include <stdint.h>

#define NB 8
#define NC 512
#define NL 2048
#define NDK 64

typedef __attribute__((ext_vector_type(16))) __bf16 v16bf;
typedef __attribute__((ext_vector_type(8)))  float  v8f;

union FragBF {
  v16bf v;
  uint4 q[2];
  __bf16 h[16];
};

__device__ inline v8f wmma_bf16(v16bf a, v16bf b, v8f c) {
  // 8 args: (neg_a, A, neg_b, B, c_mod, C, reuse_a, reuse_b)
  return __builtin_amdgcn_wmma_f32_16x16x32_bf16(false, a, false, b, (short)0, c,
                                                 false, false);
}

// ---- Fragment loaders (wave32; contiguous K order k = 16*(lane/16) + s, same for A and B) ----
// A fragment 16x32 bf16: lane&15 -> row M; lanes<16 hold K=k0..k0+15,
// lanes>=16 hold K+16. Two b128 loads per lane (source row-major, K contiguous).
__device__ inline v16bf ldfragA(const __bf16* __restrict__ P, int ld, int m0, int k0) {
  int lane = threadIdx.x & 31;
  const __bf16* p = P + (size_t)(m0 + (lane & 15)) * ld + (k0 + ((lane >> 4) << 4));
  FragBF f;
  f.q[0] = *(const uint4*)p;
  f.q[1] = *(const uint4*)(p + 8);
  return f.v;
}

// B fragment 32x16 bf16: lane&15 -> col N; lanes<16 hold K=k0..k0+15, lanes>=16 K+16.
// Source stored as B^T row-major (N-major, K contiguous) -> two b128 loads per lane.
__device__ inline v16bf ldfragB(const __bf16* __restrict__ P, int ld, int n0, int k0) {
  int lane = threadIdx.x & 31;
  const __bf16* p = P + (size_t)(n0 + (lane & 15)) * ld + (k0 + ((lane >> 4) << 4));
  FragBF f;
  f.q[0] = *(const uint4*)p;
  f.q[1] = *(const uint4*)(p + 8);
  return f.v;
}

// B fragment built from an f32 source, cvt to bf16 on the fly (fallback path).
__device__ inline v16bf ldfragB_f32(const float* __restrict__ P, int ld, int n0, int k0) {
  int lane = threadIdx.x & 31;
  const float* p = P + (size_t)(n0 + (lane & 15)) * ld + (k0 + ((lane >> 4) << 4));
  FragBF f;
#pragma unroll
  for (int i = 0; i < 16; ++i) f.h[i] = (__bf16)p[i];
  return f.v;
}

// D/C tile store helpers: lane&15 -> N, (lane>>4)*8 + vgpr -> M.
__device__ inline void st_tile_f32(float* __restrict__ D, int ldd, int m0, int n0,
                                   const v8f& c) {
  int lane = threadIdx.x & 31;
  int n = n0 + (lane & 15);
  int mb = m0 + ((lane >> 4) << 3);
#pragma unroll
  for (int v = 0; v < 8; ++v) D[(size_t)(mb + v) * ldd + n] = c[v];
}

__device__ inline void st_tile_bf(__bf16* __restrict__ D, int ldd, int m0, int n0,
                                  const v8f& c, const float* __restrict__ bias,
                                  int bias_per_row) {
  int lane = threadIdx.x & 31;
  int n = n0 + (lane & 15);
  int mb = m0 + ((lane >> 4) << 3);
#pragma unroll
  for (int v = 0; v < 8; ++v) {
    float val = c[v] + (bias_per_row ? bias[mb + v] : bias[n]);
    D[(size_t)(mb + v) * ldd + n] = (__bf16)val;
  }
}

// ---- Phase 0: transpose x [B,C,L] f32 -> xT [B,L,C] bf16 ----
__global__ void k_transpose_cvt(const float* __restrict__ x, __bf16* __restrict__ xT) {
  __shared__ float tile[32][33];
  int b  = blockIdx.z;
  int c0 = blockIdx.y * 32;
  int l0 = blockIdx.x * 32;
  const float* xb = x + (size_t)b * NC * NL;
  __bf16* xTb = xT + (size_t)b * NL * NC;
#pragma unroll
  for (int i = 0; i < 32; i += 8)
    tile[threadIdx.y + i][threadIdx.x] =
        xb[(size_t)(c0 + threadIdx.y + i) * NL + (l0 + threadIdx.x)];
  __syncthreads();
#pragma unroll
  for (int i = 0; i < 32; i += 8)
    xTb[(size_t)(l0 + threadIdx.y + i) * NC + (c0 + threadIdx.x)] =
        (__bf16)tile[threadIdx.x][threadIdx.y + i];
}

__global__ void k_cvt_bf16(const float* __restrict__ src, __bf16* __restrict__ dst, int n) {
  int i = blockIdx.x * blockDim.x + threadIdx.x;
  if (i < n) dst[i] = (__bf16)src[i];
}

// ---- Phase 1: 2x2 register-blocked NT GEMM with bias ----
// D[m,n] = sum_k A[m,k]*B[n,k] + bias ; each wave owns a 32x32 output block.
__global__ void k_nt_gemm_bias2(const __bf16* __restrict__ A, int lda, size_t sA,
                                const __bf16* __restrict__ Bm, int ldb, size_t sB,
                                __bf16* __restrict__ D, int ldd, size_t sD,
                                const float* __restrict__ bias, int bias_per_row, int K) {
  int wave = threadIdx.x >> 5;
  int b = blockIdx.z;
  int m0 = (blockIdx.x * 8 + wave) * 32;
  int n0 = blockIdx.y * 32;
  const __bf16* Ab = A + (size_t)b * sA;
  const __bf16* Bb = Bm + (size_t)b * sB;
  __bf16* Db = D + (size_t)b * sD;
  v8f c00 = {}, c01 = {}, c10 = {}, c11 = {};
  for (int k0 = 0; k0 < K; k0 += 32) {
    v16bf a0 = ldfragA(Ab, lda, m0, k0);
    v16bf a1 = ldfragA(Ab, lda, m0 + 16, k0);
    v16bf b0 = ldfragB(Bb, ldb, n0, k0);
    v16bf b1 = ldfragB(Bb, ldb, n0 + 16, k0);
    c00 = wmma_bf16(a0, b0, c00);
    c01 = wmma_bf16(a0, b1, c01);
    c10 = wmma_bf16(a1, b0, c10);
    c11 = wmma_bf16(a1, b1, c11);
  }
  st_tile_bf(Db, ldd, m0,      n0,      c00, bias, bias_per_row);
  st_tile_bf(Db, ldd, m0,      n0 + 16, c01, bias, bias_per_row);
  st_tile_bf(Db, ldd, m0 + 16, n0,      c10, bias, bias_per_row);
  st_tile_bf(Db, ldd, m0 + 16, n0 + 16, c11, bias, bias_per_row);
}

// ---- Phase 2+3 fused: energy + causal mask + softmax, final att written once ----
// One workgroup (8 waves) per 16-row block. Wave w owns j-tiles {w, w+8, ..., w+120}
// in registers (v8f acc[16], fixed-trip unrolled loops -> stays in VGPRs).
// Tiles above the diagonal are skipped (acc stays 0) and the per-element predicate
// n <= i excludes masked entries from max/sum and writes exact post-softmax zeros.
__global__ void k_energy_softmax(const __bf16* __restrict__ qbf,
                                 const __bf16* __restrict__ kbf,
                                 float* __restrict__ att, __bf16* __restrict__ attbf) {
  int w    = threadIdx.x >> 5;
  int lane = threadIdx.x & 31;
  int b    = blockIdx.y;
  int ti   = blockIdx.x;
  int i0   = ti * 16;
  const __bf16* qb = qbf + ((size_t)b * NL + i0) * NDK;
  const __bf16* kb = kbf + (size_t)b * NL * NDK;

  // q fragments loaded once, reused across all j-tiles of this wave
  v16bf a0 = ldfragA(qb, NDK, 0, 0);
  v16bf a1 = ldfragA(qb, NDK, 0, 32);

  v8f acc[16];
#pragma unroll
  for (int s = 0; s < 16; ++s) acc[s] = (v8f){};

#pragma unroll
  for (int s = 0; s < 16; ++s) {
    int jt = w + 8 * s;
    if (jt <= ti) {                 // wave-uniform causal skip
      int j0 = jt * 16;
      acc[s] = wmma_bf16(a0, ldfragB(kb, NDK, j0, 0),  acc[s]);
      acc[s] = wmma_bf16(a1, ldfragB(kb, NDK, j0, 32), acc[s]);
    }
  }

  int nn = lane & 15;               // column offset within a tile
  int mb = (lane >> 4) << 3;        // row group base (0 or 8)

  __shared__ float lred[16][9];
  __shared__ float fin[16];

  // ---- row max: per-lane partials over this wave's columns ----
  float pmax[8];
#pragma unroll
  for (int v = 0; v < 8; ++v) pmax[v] = -__builtin_inff();
#pragma unroll
  for (int s = 0; s < 16; ++s) {
    int n = (w + 8 * s) * 16 + nn;
#pragma unroll
    for (int v = 0; v < 8; ++v)
      if (n <= i0 + mb + v) pmax[v] = fmaxf(pmax[v], acc[s][v]);
  }
#pragma unroll
  for (int off = 1; off < 16; off <<= 1)     // reduce within 16-lane half-wave
#pragma unroll
    for (int v = 0; v < 8; ++v)
      pmax[v] = fmaxf(pmax[v], __shfl_xor(pmax[v], off, 32));
  if (nn == 0) {                    // lanes 0 (rows 0-7) and 16 (rows 8-15)
#pragma unroll
    for (int v = 0; v < 8; ++v) lred[mb + v][w] = pmax[v];
  }
  __syncthreads();
  if (threadIdx.x < 16) {
    float m = lred[threadIdx.x][0];
#pragma unroll
    for (int ww = 1; ww < 8; ++ww) m = fmaxf(m, lred[threadIdx.x][ww]);
    fin[threadIdx.x] = m;
  }
  __syncthreads();
  float rmax[8];
#pragma unroll
  for (int v = 0; v < 8; ++v) rmax[v] = fin[mb + v];
  __syncthreads();

  // ---- exp + row sum (predicated); exp values kept in acc ----
  float psum[8];
#pragma unroll
  for (int v = 0; v < 8; ++v) psum[v] = 0.f;
#pragma unroll
  for (int s = 0; s < 16; ++s) {
    int n = (w + 8 * s) * 16 + nn;
#pragma unroll
    for (int v = 0; v < 8; ++v) {
      float e = __expf(acc[s][v] - rmax[v]);
      e = (n <= i0 + mb + v) ? e : 0.f;
      acc[s][v] = e;
      psum[v] += e;
    }
  }
#pragma unroll
  for (int off = 1; off < 16; off <<= 1)
#pragma unroll
    for (int v = 0; v < 8; ++v)
      psum[v] += __shfl_xor(psum[v], off, 32);
  if (nn == 0) {
#pragma unroll
    for (int v = 0; v < 8; ++v) lred[mb + v][w] = psum[v];
  }
  __syncthreads();
  if (threadIdx.x < 16) {
    float ssum = 0.f;
#pragma unroll
    for (int ww = 0; ww < 8; ++ww) ssum += lred[threadIdx.x][ww];
    fin[threadIdx.x] = 1.0f / ssum;
  }
  __syncthreads();
  float rinv[8];
#pragma unroll
  for (int v = 0; v < 8; ++v) rinv[v] = fin[mb + v];

  // ---- normalize and write final attention (f32 + optional bf16 mirror) ----
  float* attb = att + ((size_t)b * NL + i0) * NL;
  __bf16* atbb = attbf ? attbf + ((size_t)b * NL + i0) * NL : nullptr;
#pragma unroll
  for (int s = 0; s < 16; ++s) {
    int n = (w + 8 * s) * 16 + nn;
#pragma unroll
    for (int v = 0; v < 8; ++v) {
      float r = acc[s][v] * rinv[v];
      attb[(size_t)(mb + v) * NL + n] = r;
      if (atbb) atbb[(size_t)(mb + v) * NL + n] = (__bf16)r;
    }
  }
}

// ---- Phase 4a: out = V x Att^T with bf16 attention (preferred), 2x2 blocked ----
__global__ void k_out_bf(const __bf16* __restrict__ vbf, const __bf16* __restrict__ attbf,
                         float* __restrict__ out) {
  int wave = threadIdx.x >> 5;
  int b  = blockIdx.z;
  int m0 = (blockIdx.x * 8 + wave) * 32;   // channel block
  int n0 = blockIdx.y * 32;                // position block
  const __bf16* vb = vbf + (size_t)b * NC * NL;
  const __bf16* ab = attbf + (size_t)b * NL * NL;
  v8f c00 = {}, c01 = {}, c10 = {}, c11 = {};
  int kmax = n0 + 32;                      // causal: j <= i < n0+32; tail is exact zeros
  for (int k0 = 0; k0 < kmax; k0 += 32) {
    v16bf a0 = ldfragA(vb, NL, m0, k0);
    v16bf a1 = ldfragA(vb, NL, m0 + 16, k0);
    v16bf b0 = ldfragB(ab, NL, n0, k0);
    v16bf b1 = ldfragB(ab, NL, n0 + 16, k0);
    c00 = wmma_bf16(a0, b0, c00);
    c01 = wmma_bf16(a0, b1, c01);
    c10 = wmma_bf16(a1, b0, c10);
    c11 = wmma_bf16(a1, b1, c11);
  }
  float* ob = out + (size_t)b * NC * NL;
  st_tile_f32(ob, NL, m0,      n0,      c00);
  st_tile_f32(ob, NL, m0,      n0 + 16, c01);
  st_tile_f32(ob, NL, m0 + 16, n0,      c10);
  st_tile_f32(ob, NL, m0 + 16, n0 + 16, c11);
}

// ---- Phase 4b: fallback with f32 attention + on-the-fly cvt (small ws) ----
__global__ void k_out_f32(const __bf16* __restrict__ vbf, const float* __restrict__ att,
                          float* __restrict__ out) {
  int wave = threadIdx.x >> 5;
  int b  = blockIdx.z;
  int m0 = (blockIdx.x * 8 + wave) * 32;
  int n0 = blockIdx.y * 32;
  const __bf16* vb = vbf + (size_t)b * NC * NL;
  const float* ab = att + (size_t)b * NL * NL;
  v8f c00 = {}, c01 = {}, c10 = {}, c11 = {};
  int kmax = n0 + 32;
  for (int k0 = 0; k0 < kmax; k0 += 32) {
    v16bf a0 = ldfragA(vb, NL, m0, k0);
    v16bf a1 = ldfragA(vb, NL, m0 + 16, k0);
    v16bf b0 = ldfragB_f32(ab, NL, n0, k0);
    v16bf b1 = ldfragB_f32(ab, NL, n0 + 16, k0);
    c00 = wmma_bf16(a0, b0, c00);
    c01 = wmma_bf16(a0, b1, c01);
    c10 = wmma_bf16(a1, b0, c10);
    c11 = wmma_bf16(a1, b1, c11);
  }
  float* ob = out + (size_t)b * NC * NL;
  st_tile_f32(ob, NL, m0,      n0,      c00);
  st_tile_f32(ob, NL, m0,      n0 + 16, c01);
  st_tile_f32(ob, NL, m0 + 16, n0,      c10);
  st_tile_f32(ob, NL, m0 + 16, n0 + 16, c11);
}

extern "C" void kernel_launch(void* const* d_in, const int* in_sizes, int n_in,
                              void* d_out, int out_size, void* d_ws, size_t ws_size,
                              hipStream_t stream) {
  const float* x  = (const float*)d_in[0];
  const float* Wq = (const float*)d_in[1];
  const float* bq = (const float*)d_in[2];
  const float* Wk = (const float*)d_in[3];
  const float* bk = (const float*)d_in[4];
  const float* Wv = (const float*)d_in[5];
  const float* bv = (const float*)d_in[6];

  float* out = (float*)d_out;
  float* att = out + (size_t)NB * NC * NL;   // tuple order: (out, attention)

  // Workspace carve-up (bf16 elements, all offsets 16B-aligned)
  __bf16* xT  = (__bf16*)d_ws;                         // [B, L, C]   8,388,608
  __bf16* wqb = xT  + (size_t)NB * NL * NC;            // [DK, C]     32,768
  __bf16* wkb = wqb + (size_t)NDK * NC;                // [DK, C]     32,768
  __bf16* wvb = wkb + (size_t)NDK * NC;                // [C, C]      262,144
  __bf16* qb  = wvb + (size_t)NC * NC;                 // [B, L, DK]  1,048,576
  __bf16* ktb = qb  + (size_t)NB * NL * NDK;           // [B, L, DK]  1,048,576
  __bf16* vb  = ktb + (size_t)NB * NL * NDK;           // [B, C, L]   8,388,608
  __bf16* atb = vb  + (size_t)NB * NC * NL;            // [B, L, L]   33,554,432 (optional)

  size_t base_elems = (size_t)NB * NL * NC + 2u * NDK * NC + (size_t)NC * NC +
                      2u * (size_t)NB * NL * NDK + (size_t)NB * NC * NL;
  size_t full_elems = base_elems + (size_t)NB * NL * NL;
  bool use_attbf = ws_size >= full_elems * sizeof(__bf16);  // ws_size fixed -> deterministic

  // Phase 0: transpose/convert activations + convert weights
  k_transpose_cvt<<<dim3(NL / 32, NC / 32, NB), dim3(32, 8), 0, stream>>>(x, xT);
  k_cvt_bf16<<<(NDK * NC + 255) / 256, 256, 0, stream>>>(Wq, wqb, NDK * NC);
  k_cvt_bf16<<<(NDK * NC + 255) / 256, 256, 0, stream>>>(Wk, wkb, NDK * NC);
  k_cvt_bf16<<<(NC * NC + 255) / 256, 256, 0, stream>>>(Wv, wvb, NC * NC);

  // Phase 1: projections (2x2-blocked NT GEMMs, K = C = 512)
  k_nt_gemm_bias2<<<dim3(8, 2, NB), 256, 0, stream>>>(
      xT, NC, (size_t)NL * NC, wqb, NC, 0, qb, NDK, (size_t)NL * NDK, bq, 0, NC);
  k_nt_gemm_bias2<<<dim3(8, 2, NB), 256, 0, stream>>>(
      xT, NC, (size_t)NL * NC, wkb, NC, 0, ktb, NDK, (size_t)NL * NDK, bk, 0, NC);
  k_nt_gemm_bias2<<<dim3(2, 64, NB), 256, 0, stream>>>(
      wvb, NC, 0, xT, NC, (size_t)NL * NC, vb, NL, (size_t)NC * NL, bv, 1, NC);

  // Phase 2+3 fused: energy + causal mask + softmax -> final attention in one pass
  k_energy_softmax<<<dim3(NL / 16, NB), 256, 0, stream>>>(
      qb, ktb, att, use_attbf ? atb : nullptr);

  // Phase 4: out = V x Att^T (causally truncated K), 2x2 blocked
  if (use_attbf) {
    k_out_bf<<<dim3(2, 64, NB), 256, 0, stream>>>(vb, atb, out);
  } else {
    k_out_f32<<<dim3(2, 64, NB), 256, 0, stream>>>(vb, att, out);
  }
}
